// RelLearnableDecoderLayer_55319178773049
// MI455X (gfx1250) — matrix operand order repaired and checked by reading, not compiled, not run on hardware
//
#include <hip/hip_runtime.h>

// ---------------------------------------------------------------------------
// CDNA5 (gfx1250) Transformer-XL RelLearnableDecoderLayer
// QLEN=1024, BSZ=4, D_MODEL=1024, N_HEAD=16, D_HEAD=64, D_INNER=4096
// Matmuls via v_wmma_f32_16x16x32_bf16; GEMM LDS staging via the Tensor Data
// Mover (tensor_load_to_lds + s_wait_tensorcnt) when the builtin exists.
// ---------------------------------------------------------------------------

typedef __attribute__((ext_vector_type(16))) __bf16 v16bf;
typedef __attribute__((ext_vector_type(2)))  __bf16 v2bf;
typedef __attribute__((ext_vector_type(8)))  float  v8f;
typedef __attribute__((ext_vector_type(4)))  unsigned uv4;
typedef __attribute__((ext_vector_type(8)))  int      iv8;
typedef __attribute__((ext_vector_type(4)))  int      iv4;

union FragAB { v16bf v; uint4 q[2]; };
union FragC  { v8f v; float e[8]; };

#if __has_builtin(__builtin_amdgcn_cvt_pk_bf16_f32)
static __device__ __forceinline__ unsigned pk2(float a, float b) {
  union { v2bf v; unsigned u; } cv;
  cv.v = __builtin_amdgcn_cvt_pk_bf16_f32(a, b);
  return cv.u;
}
static __device__ __forceinline__ unsigned short f2bf_u(float x) {
  return (unsigned short)pk2(x, x);
}
#else
// native __bf16 conversion: selects v_cvt_pk_bf16_f32 on gfx1250
static __device__ __forceinline__ unsigned short f2bf_u(float x) {
  __bf16 h = (__bf16)x;
  return __builtin_bit_cast(unsigned short, h);
}
static __device__ __forceinline__ unsigned pk2(float a, float b) {
  union { v2bf v; unsigned u; } cv;
  cv.v.x = (__bf16)a; cv.v.y = (__bf16)b;
  return cv.u;
}
#endif

static __device__ __forceinline__ float bf2f(unsigned short h) {
  return __uint_as_float(((unsigned)h) << 16);
}
static __device__ __forceinline__ v8f wmma_bf16(const FragAB& a, const FragAB& b, v8f c) {
  return __builtin_amdgcn_wmma_f32_16x16x32_bf16(false, a.v, false, b.v, (short)0, c, false, false);
}

// ---------------------------------------------------------------------------
// Tensor Data Mover staging (gfx1250). 2D tile load global->LDS with native
// LDS padding: pad_interval = 16 DWORDs (64B), pad_amount = 4 DWORDs (16B)
// -> 80B LDS row pitch, matching the WMMA fragment loaders below.
// ---------------------------------------------------------------------------
#if __has_builtin(__builtin_amdgcn_tensor_load_to_lds) && __has_builtin(__builtin_amdgcn_s_wait_tensorcnt)
#define GEMM_USE_TDM 1
#else
#define GEMM_USE_TDM 0
#endif

#if GEMM_USE_TDM
static __device__ __forceinline__ unsigned lds_off(const void* p) {
  return (unsigned)(size_t)p;   // flat LDS address: offset in low 32 bits
}
// tile: tile_rows x 64B (8 units of 8B); row stride = ld elements * 2B
static __device__ __forceinline__ void tdm_load_tile(
    unsigned ldsoff, const unsigned short* g, unsigned K_elems,
    unsigned rows_total, unsigned ld_elems, unsigned tile_rows) {
  const unsigned long long ga = (unsigned long long)(size_t)g;
  const unsigned td0 = K_elems >> 2;       // row length, 8B units
  const unsigned td1 = rows_total;         // tensor rows
  const unsigned st0 = ld_elems >> 2;      // row stride, 8B units
  union { uv4 v; unsigned u[4]; } g0;
  g0.u[0] = 1u;                                   // count=1, user mode
  g0.u[1] = ldsoff;                               // lds_addr
  g0.u[2] = (unsigned)ga;                         // global_addr[31:0]
  g0.u[3] = (unsigned)(ga >> 32) | (2u << 30);    // global_addr[56:32] | type=2
  union { iv8 v; unsigned u[8]; } g1;
  g1.u[0] = (3u << 16) | (1u << 20) | (3u << 22) | (3u << 25); // 8B elems, pad 64B->+16B
  g1.u[1] = (td0 & 0xffffu) << 16;                // tensor_dim0[15:0]
  g1.u[2] = (td0 >> 16) | ((td1 & 0xffffu) << 16);// tensor_dim0[31:16] | tensor_dim1[15:0]
  g1.u[3] = (td1 >> 16) | (8u << 16);             // tensor_dim1[31:16] | tile_dim0=8
  g1.u[4] = tile_rows;                            // tile_dim1, tile_dim2=0
  g1.u[5] = st0;                                  // tensor_dim0_stride[31:0]
  g1.u[6] = 0u;                                   // stride0 hi | stride1 lo
  g1.u[7] = 0u;
  iv4 z4 = {0, 0, 0, 0};
#if __clang_major__ >= 23
  iv8 z8 = {0, 0, 0, 0, 0, 0, 0, 0};
  __builtin_amdgcn_tensor_load_to_lds(g0.v, g1.v, z4, z4, z8, 0);
#else
  __builtin_amdgcn_tensor_load_to_lds(g0.v, g1.v, z4, z4, 0);
#endif
}
#endif  // GEMM_USE_TDM

// ---------------------------------------------------------------------------
// Elementwise f32 -> bf16 (vectorized, packed convert)
// ---------------------------------------------------------------------------
__global__ __launch_bounds__(256) void cvt_bf16_kernel(
    const float* __restrict__ in, unsigned short* __restrict__ out, int n4) {
  const int id = blockIdx.x * 256 + threadIdx.x;
  if (id < n4) {
    const float4 v = ((const float4*)in)[id];
    uint2 o;
    o.x = pk2(v.x, v.y);
    o.y = pk2(v.z, v.w);
    ((uint2*)out)[id] = o;
  }
}

// ---------------------------------------------------------------------------
// NT GEMM, bf16 x bf16 -> f32 acc: C[M,N] = A[M,K] * Bw[N,K]^T
// 256 threads = 8 waves (2Mx4N); block tile 128x256; wave tile 64x64
// (16 WMMA per k-step from 8 fragment loads -> 1.0 ds_load_b128 per WMMA).
// k-step 32; double-buffered LDS; 80B row pitch (conflict-free b128 loads).
// Staging: TDM descriptors (wave 0) when available, else load/store copy.
// ---------------------------------------------------------------------------
template <bool BIAS, bool RELU, bool OUTBF>
__global__ __launch_bounds__(256) void gemm_bb(
    const unsigned short* __restrict__ A, const unsigned short* __restrict__ Bw,
    const float* __restrict__ bias, void* __restrict__ Cp,
    int M, int N, int K, int lda, int ldb, int ldc) {
  constexpr int BM = 128, BN = 256, KS = 32;
  __shared__ __attribute__((aligned(16))) unsigned short sA[2][BM * 40];
  __shared__ __attribute__((aligned(16))) unsigned short sB[2][BN * 40];
  const int tid = threadIdx.x;
  const int lane = tid & 31, w = tid >> 5;
  const int nn = lane & 15, lh = lane >> 4;
  const int m0 = blockIdx.y * BM, n0 = blockIdx.x * BN;
  const int waveM = w >> 2, waveN = w & 3;   // 2 x 4 wave grid
  const int nk = K / KS;

  FragC acc[4][4];
#pragma unroll
  for (int rt = 0; rt < 4; ++rt)
#pragma unroll
    for (int cn = 0; cn < 4; ++cn)
#pragma unroll
      for (int r = 0; r < 8; ++r) acc[rt][cn].e[r] = 0.f;

#if GEMM_USE_TDM
  if (w == 0) {
    tdm_load_tile(lds_off(&sA[0][0]), A + (size_t)m0 * lda, (unsigned)K, (unsigned)M, (unsigned)lda, BM);
    tdm_load_tile(lds_off(&sB[0][0]), Bw + (size_t)n0 * ldb, (unsigned)K, (unsigned)N, (unsigned)ldb, BN);
    __builtin_amdgcn_s_wait_tensorcnt(0);
  }
  __syncthreads();
#else
  auto stage = [&](int buf, int k0) {
#pragma unroll
    for (int it = 0; it < 4; ++it) {           // A: 1024 chunks of 4 elems
      const int c = tid + it * 256;
      const int row = c >> 3, kq = (c & 7) << 2;
      const uint2 ua = *(const uint2*)(A + (size_t)(m0 + row) * lda + k0 + kq);
      unsigned* da = (unsigned*)&sA[buf][0] + row * 20 + (kq >> 1);
      da[0] = ua.x; da[1] = ua.y;
    }
#pragma unroll
    for (int it = 0; it < 8; ++it) {           // B: 2048 chunks of 4 elems
      const int c = tid + it * 256;
      const int row = c >> 3, kq = (c & 7) << 2;
      const uint2 ub = *(const uint2*)(Bw + (size_t)(n0 + row) * ldb + k0 + kq);
      unsigned* db = (unsigned*)&sB[buf][0] + row * 20 + (kq >> 1);
      db[0] = ub.x; db[1] = ub.y;
    }
  };
  stage(0, 0);
#endif

  for (int kt = 0; kt < nk; ++kt) {
#if GEMM_USE_TDM
    if (w == 0 && kt + 1 < nk) {
      const int k0 = (kt + 1) * KS, buf = (kt + 1) & 1;
      tdm_load_tile(lds_off(&sA[buf][0]), A + (size_t)m0 * lda + k0, (unsigned)K, (unsigned)M, (unsigned)lda, BM);
      tdm_load_tile(lds_off(&sB[buf][0]), Bw + (size_t)n0 * ldb + k0, (unsigned)K, (unsigned)N, (unsigned)ldb, BN);
    }
#else
    __syncthreads();
    if (kt + 1 < nk) stage((kt + 1) & 1, (kt + 1) * KS);
#endif
    const int buf = kt & 1;
    FragAB afr[4];
#pragma unroll
    for (int rt = 0; rt < 4; ++rt) {
      const char* ab = (const char*)&sA[buf][0] + (waveM * 64 + rt * 16 + nn) * 80 + lh * 16;
      afr[rt].q[0] = *(const uint4*)(ab);
      afr[rt].q[1] = *(const uint4*)(ab + 32);
    }
#pragma unroll
    for (int cn = 0; cn < 4; ++cn) {
      FragAB bf;
      const char* bb = (const char*)&sB[buf][0] + (waveN * 64 + cn * 16 + nn) * 80 + lh * 16;
      bf.q[0] = *(const uint4*)(bb);
      bf.q[1] = *(const uint4*)(bb + 32);
#pragma unroll
      for (int rt = 0; rt < 4; ++rt)
        acc[rt][cn].v = wmma_bf16(afr[rt], bf, acc[rt][cn].v);
    }
#if GEMM_USE_TDM
    if (w == 0) __builtin_amdgcn_s_wait_tensorcnt(0);
    __syncthreads();
#endif
  }
  // epilogue
#pragma unroll
  for (int rt = 0; rt < 4; ++rt)
#pragma unroll
    for (int cn = 0; cn < 4; ++cn) {
      const int gn = n0 + waveN * 64 + cn * 16 + nn;
      float bv = 0.f;
      if constexpr (BIAS) bv = bias[gn];
#pragma unroll
      for (int r = 0; r < 8; ++r) {
        const int gm = m0 + waveM * 64 + rt * 16 + r + 8 * lh;
        float vv = acc[rt][cn].e[r] + bv;
        if constexpr (RELU) vv = fmaxf(vv, 0.f);
        if constexpr (OUTBF)
          ((unsigned short*)Cp)[(size_t)gm * ldc + gn] = f2bf_u(vv);
        else
          ((float*)Cp)[(size_t)gm * ldc + gn] = vv;
      }
    }
}

// ---------------------------------------------------------------------------
// corr[n][c] = r_w_bias[n,:] . r_emb[c,n,:]   (rel-shift correction term)
// ---------------------------------------------------------------------------
__global__ __launch_bounds__(256) void corr_kernel(
    const float* __restrict__ r_w_bias, const float* __restrict__ r_emb,
    float* __restrict__ corr) {
  const int id = blockIdx.x * 256 + threadIdx.x;   // 16384 = 1024*16
  const int nh = id & 15, c = id >> 4;
  float s = 0.f;
#pragma unroll 8
  for (int d = 0; d < 64; ++d)
    s += r_w_bias[nh * 64 + d] * r_emb[((size_t)c * 16 + nh) * 64 + d];
  corr[nh * 1024 + c] = s;
}

// ---------------------------------------------------------------------------
// Causal flash attention with Transformer-XL relative bias (bf16 inputs).
// Grid: (QLEN/64, BSZ*N_HEAD). 128 threads = 4 waves x 16 query rows.
// For j<=i: score = (q+rwb).k[j] + (q+rwb).r_emb[c] + (r_bias[c]-corr[c]),
//           c = QLEN-1 + j - i.  Bias band computed with WMMA, realigned
//           per diagonal through a per-wave LDS tile (sT).
// ---------------------------------------------------------------------------
__global__ __launch_bounds__(128) void flash_rel_attn(
    const unsigned short* __restrict__ headsb, const unsigned short* __restrict__ rembb,
    const float* __restrict__ r_w_bias, const float* __restrict__ r_bias,
    const float* __restrict__ corr, unsigned short* __restrict__ avecb) {
  constexpr int QL = 1024, BSZ = 4, NH = 16, BI = 64, BJ = 32;
  const int tid = threadIdx.x;
  const int lane = tid & 31, w = tid >> 5;
  const int nn = lane & 15, lh = lane >> 4;
  const int bn = blockIdx.y;
  const int b = bn & 3, n = bn >> 2;
  const int i0 = blockIdx.x * BI;
  const int iw0 = i0 + 16 * w;
  const float scale = 0.125f;   // 1/sqrt(64)

  __shared__ __attribute__((aligned(16))) unsigned short sQ[64 * 72];   // 64 rows x 144B
  __shared__ __attribute__((aligned(16))) unsigned short sK[32 * 72];   // 32 rows x 144B
  __shared__ __attribute__((aligned(16))) unsigned short sVt[64 * 40];  // 64 d x 80B (transposed V)
  __shared__ __attribute__((aligned(16))) unsigned short sR[96 * 72];   // r_emb band
  __shared__ __attribute__((aligned(16))) float sRb[96];                // r_bias - corr band
  __shared__ __attribute__((aligned(16))) unsigned short sT[4 * 16 * 48];// per-wave bias tiles (bf16)
  __shared__ __attribute__((aligned(16))) unsigned short sP[4 * 16 * 40];// per-wave P (bf16)

  // ---- stage Q (+ r_w_bias), once per block ----
  for (int c = tid; c < 64 * 16; c += 128) {
    const int row = c >> 4, d4 = (c & 15) << 2;
    const uint2 q = *(const uint2*)(headsb + ((size_t)((i0 + row) * BSZ + b)) * 3072 + n * 64 + d4);
    const float4 wv = *(const float4*)(r_w_bias + n * 64 + d4);
    unsigned* dst = (unsigned*)sQ + row * 36 + (d4 >> 1);
    dst[0] = pk2(bf2f((unsigned short)q.x) + wv.x, bf2f((unsigned short)(q.x >> 16)) + wv.y);
    dst[1] = pk2(bf2f((unsigned short)q.y) + wv.z, bf2f((unsigned short)(q.y >> 16)) + wv.w);
  }
  __syncthreads();

  FragAB qf[2];
  {
    const char* base = (const char*)sQ + (w * 16 + nn) * 144 + lh * 16;
    qf[0].q[0] = *(const uint4*)(base);       qf[0].q[1] = *(const uint4*)(base + 32);
    qf[1].q[0] = *(const uint4*)(base + 64);  qf[1].q[1] = *(const uint4*)(base + 96);
  }

  float m_r[8], l_r[8];
  FragC O[4];
#pragma unroll
  for (int r = 0; r < 8; ++r) { m_r[r] = -3.0e38f; l_r[r] = 0.f; }
#pragma unroll
  for (int o = 0; o < 4; ++o)
#pragma unroll
    for (int r = 0; r < 8; ++r) O[o].e[r] = 0.f;

  const int nblocks = (i0 + BI) / BJ;
  for (int jb = 0; jb < nblocks; ++jb) {
    const int j0 = jb * BJ;
    const int cb = j0 + QL - 1 - i0 - 63;   // band base (band end may pass QL-1; clamped below)
    __syncthreads();
    // stage K (raw bf16 copy)
    for (int c = tid; c < 32 * 16; c += 128) {
      const int row = c >> 4, d4 = (c & 15) << 2;
      const uint2 kv = *(const uint2*)(headsb + ((size_t)((j0 + row) * BSZ + b)) * 3072 + 1024 + n * 64 + d4);
      unsigned* dst = (unsigned*)sK + row * 36 + (d4 >> 1);
      dst[0] = kv.x; dst[1] = kv.y;
    }
    // stage V transposed: sVt[d][j]
    for (int c = tid; c < 32 * 16; c += 128) {
      const int row = c >> 4, d4 = (c & 15) << 2;
      const uint2 vv = *(const uint2*)(headsb + ((size_t)((j0 + row) * BSZ + b)) * 3072 + 2048 + n * 64 + d4);
      sVt[(d4 + 0) * 40 + row] = (unsigned short)vv.x;
      sVt[(d4 + 1) * 40 + row] = (unsigned short)(vv.x >> 16);
      sVt[(d4 + 2) * 40 + row] = (unsigned short)vv.y;
      sVt[(d4 + 3) * 40 + row] = (unsigned short)(vv.y >> 16);
    }
    // stage r_emb band (raw bf16 copy) + (r_bias - corr) band
    for (int c = tid; c < 96 * 16; c += 128) {
      const int row = c >> 4, d4 = (c & 15) << 2;
      int cidx = cb + row; cidx = cidx < 0 ? 0 : (cidx > QL - 1 ? QL - 1 : cidx);
      const uint2 rv = *(const uint2*)(rembb + ((size_t)cidx * NH + n) * 64 + d4);
      unsigned* dst = (unsigned*)sR + row * 36 + (d4 >> 1);
      dst[0] = rv.x; dst[1] = rv.y;
    }
    if (tid < 96) {
      int cidx = cb + tid; cidx = cidx < 0 ? 0 : (cidx > QL - 1 ? QL - 1 : cidx);
      sRb[tid] = r_bias[(size_t)cidx * NH + n] - corr[n * 1024 + cidx];
    }
    __syncthreads();
    if (j0 > iw0 + 15) continue;   // wave-uniform skip (fully masked block)

    // ---- S = (q+rwb) . k^T  (2 tiles x 2 k-chunks) ----
    FragC s[2];
#pragma unroll
    for (int c2 = 0; c2 < 2; ++c2) {
#pragma unroll
      for (int r = 0; r < 8; ++r) s[c2].e[r] = 0.f;
#pragma unroll
      for (int kc = 0; kc < 2; ++kc) {
        FragAB bk;
        const char* base = (const char*)sK + (c2 * 16 + nn) * 144 + kc * 64 + lh * 16;
        bk.q[0] = *(const uint4*)(base); bk.q[1] = *(const uint4*)(base + 32);
        s[c2].v = wmma_bf16(qf[kc], bk, s[c2].v);
      }
    }
    // ---- bias band tiles T[di, t] = (q+rwb) . r_emb[band]  -> sT ----
#pragma unroll
    for (int tt = 0; tt < 3; ++tt) {
      FragC tf;
#pragma unroll
      for (int r = 0; r < 8; ++r) tf.e[r] = 0.f;
#pragma unroll
      for (int kc = 0; kc < 2; ++kc) {
        FragAB br;
        const char* base = (const char*)sR + ((48 - 16 * w) + tt * 16 + nn) * 144 + kc * 64 + lh * 16;
        br.q[0] = *(const uint4*)(base); br.q[1] = *(const uint4*)(base + 32);
        tf.v = wmma_bf16(qf[kc], br, tf.v);
      }
#pragma unroll
      for (int r = 0; r < 8; ++r) {
        const int di = r + 8 * lh;
        sT[(w * 16 + di) * 48 + tt * 16 + nn] = f2bf_u(tf.e[r]);
      }
    }
    // ---- add diagonal-aligned bias, scale, causal mask ----
#pragma unroll
    for (int c2 = 0; c2 < 2; ++c2)
#pragma unroll
      for (int r = 0; r < 8; ++r) {
        const int di = r + 8 * lh;
        const int t = c2 * 16 + nn - di + 15;                 // 0..46
        const float bias = bf2f(sT[(w * 16 + di) * 48 + t]) + sRb[48 - 16 * w + t];
        const float val = (s[c2].e[r] + bias) * scale;
        const int j = j0 + c2 * 16 + nn, i = iw0 + di;
        s[c2].e[r] = (j > i) ? -1.0e30f : val;
      }
    // ---- online softmax update ----
    float alpha[8];
#pragma unroll
    for (int r = 0; r < 8; ++r) {
      float mv = fmaxf(s[0].e[r], s[1].e[r]);
      mv = fmaxf(mv, __shfl_xor(mv, 1));
      mv = fmaxf(mv, __shfl_xor(mv, 2));
      mv = fmaxf(mv, __shfl_xor(mv, 4));
      mv = fmaxf(mv, __shfl_xor(mv, 8));
      const float nm = fmaxf(m_r[r], mv);
      alpha[r] = __expf(m_r[r] - nm);
      m_r[r] = nm;
    }
#pragma unroll
    for (int c2 = 0; c2 < 2; ++c2)
#pragma unroll
      for (int r = 0; r < 8; ++r) s[c2].e[r] = __expf(s[c2].e[r] - m_r[r]);
#pragma unroll
    for (int r = 0; r < 8; ++r) {
      float ls = s[0].e[r] + s[1].e[r];
      ls += __shfl_xor(ls, 1);
      ls += __shfl_xor(ls, 2);
      ls += __shfl_xor(ls, 4);
      ls += __shfl_xor(ls, 8);
      l_r[r] = l_r[r] * alpha[r] + ls;
    }
#pragma unroll
    for (int o = 0; o < 4; ++o)
#pragma unroll
      for (int r = 0; r < 8; ++r) O[o].e[r] *= alpha[r];
    // ---- P -> bf16 LDS -> A-frag; O += P.V ----
#pragma unroll
    for (int c2 = 0; c2 < 2; ++c2)
#pragma unroll
      for (int r = 0; r < 8; ++r) {
        const int di = r + 8 * lh;
        sP[(w * 16 + di) * 40 + c2 * 16 + nn] = f2bf_u(s[c2].e[r]);
      }
    FragAB pf;
    {
      const char* base = (const char*)sP + (w * 16 + nn) * 80 + lh * 16;
      pf.q[0] = *(const uint4*)(base); pf.q[1] = *(const uint4*)(base + 32);
    }
#pragma unroll
    for (int o = 0; o < 4; ++o) {
      FragAB vb;
      const char* base = (const char*)sVt + (o * 16 + nn) * 80 + lh * 16;
      vb.q[0] = *(const uint4*)(base); vb.q[1] = *(const uint4*)(base + 32);
      O[o].v = wmma_bf16(pf, vb, O[o].v);
    }
  }

  // ---- normalize and write attn_vec (bf16) [(i*BSZ+b), n*64+d] ----
#pragma unroll
  for (int r = 0; r < 8; ++r) l_r[r] = 1.0f / l_r[r];
#pragma unroll
  for (int o = 0; o < 4; ++o)
#pragma unroll
    for (int r = 0; r < 8; ++r) {
      const int di = r + 8 * lh;
      const int i = iw0 + di;
      avecb[((size_t)(i * BSZ + b) * NH + n) * 64 + o * 16 + nn] = f2bf_u(O[o].e[r] * l_r[r]);
    }
}

// ---------------------------------------------------------------------------
// out = LayerNorm(x1 + x2) * g + b  over rows of 1024 (optional bf16 mirror)
// ---------------------------------------------------------------------------
template <bool BFOUT>
__global__ __launch_bounds__(256) void ln_residual(
    const float* __restrict__ x1, const float* __restrict__ x2,
    const float* __restrict__ g, const float* __restrict__ bt,
    float* __restrict__ out, unsigned short* __restrict__ outb) {
  const int row = blockIdx.x, tid = threadIdx.x;
  __shared__ float red[8];
  float v[4];
  float p = 0.f;
#pragma unroll
  for (int t = 0; t < 4; ++t) {
    const int c = tid + t * 256;
    v[t] = x1[(size_t)row * 1024 + c] + x2[(size_t)row * 1024 + c];
    p += v[t];
  }
  for (int m = 1; m < 32; m <<= 1) p += __shfl_xor(p, m);
  if ((tid & 31) == 0) red[tid >> 5] = p;
  __syncthreads();
  float tot = 0.f;
#pragma unroll
  for (int i = 0; i < 8; ++i) tot += red[i];
  const float mu = tot * (1.f / 1024.f);
  __syncthreads();
  p = 0.f;
#pragma unroll
  for (int t = 0; t < 4; ++t) { const float d = v[t] - mu; p += d * d; }
  for (int m = 1; m < 32; m <<= 1) p += __shfl_xor(p, m);
  if ((tid & 31) == 0) red[tid >> 5] = p;
  __syncthreads();
  tot = 0.f;
#pragma unroll
  for (int i = 0; i < 8; ++i) tot += red[i];
  const float rs = rsqrtf(tot * (1.f / 1024.f) + 1e-5f);
#pragma unroll
  for (int t = 0; t < 4; ++t) {
    const int c = tid + t * 256;
    const float y = (v[t] - mu) * rs * g[c] + bt[c];
    out[(size_t)row * 1024 + c] = y;
    if constexpr (BFOUT) outb[(size_t)row * 1024 + c] = f2bf_u(y);
  }
}

// ---------------------------------------------------------------------------
extern "C" void kernel_launch(void* const* d_in, const int* in_sizes, int n_in,
                              void* d_out, int out_size, void* d_ws, size_t ws_size,
                              hipStream_t stream) {
  (void)in_sizes; (void)n_in; (void)out_size; (void)ws_size;
  const float* dec_inp  = (const float*)d_in[0];
  const float* r_emb    = (const float*)d_in[1];
  const float* r_w_bias = (const float*)d_in[2];
  const float* r_bias   = (const float*)d_in[3];
  // d_in[4] = dec_attn_mask (causal triu; structure exploited directly)
  const float* qkv_w    = (const float*)d_in[5];
  const float* o_w      = (const float*)d_in[6];
  const float* ln1_g    = (const float*)d_in[7];
  const float* ln1_b    = (const float*)d_in[8];
  const float* ff_w1    = (const float*)d_in[9];
  const float* ff_b1    = (const float*)d_in[10];
  const float* ff_w2    = (const float*)d_in[11];
  const float* ff_b2    = (const float*)d_in[12];
  const float* ln2_g    = (const float*)d_in[13];
  const float* ln2_b    = (const float*)d_in[14];

  char* ws = (char*)d_ws;
  unsigned short* headsb = (unsigned short*)(ws);                  // 4096x3072 bf16 (24MB)
  unsigned short* avecb  = (unsigned short*)(ws +  25165824ull);   // 4096x1024 bf16 ( 8MB)
  float*          aout   = (float*)         (ws +  33554432ull);   // 4096x1024 f32 (16MB)
  float*          out1   = (float*)         (ws +  50331648ull);   // 4096x1024 f32 (16MB)
  unsigned short* out1b  = (unsigned short*)(ws +  67108864ull);   // 4096x1024 bf16 ( 8MB)
  unsigned short* hbuf   = (unsigned short*)(ws +  75497472ull);   // 4096x4096 bf16 (32MB)
  float*          corr   = (float*)         (ws + 109051904ull);   // 16x1024 f32 (64KB)
  unsigned short* qkv_b  = (unsigned short*)(ws + 109117440ull);   // 3072x1024 bf16 (6MB)
  unsigned short* ow_b   = (unsigned short*)(ws + 115408896ull);   // 1024x1024 bf16 (2MB)
  unsigned short* ff1_b  = (unsigned short*)(ws + 117506048ull);   // 4096x1024 bf16 (8MB)
  unsigned short* ff2_b  = (unsigned short*)(ws + 125894656ull);   // 1024x4096 bf16 (8MB)
  unsigned short* din_b  = (unsigned short*)(ws + 134283264ull);   // 4096x1024 bf16 (8MB)
  unsigned short* remb_b = (unsigned short*)(ws + 142671872ull);   // 1024x16x64 bf16 (2MB)
  float*          core   = (float*)ws;                             // reuse heads region (f32)

  // --- pre-convert weights / static inputs to bf16 ---
  cvt_bf16_kernel<<<3072, 256, 0, stream>>>(qkv_w,  qkv_b,  786432);
  cvt_bf16_kernel<<<1024, 256, 0, stream>>>(o_w,    ow_b,   262144);
  cvt_bf16_kernel<<<4096, 256, 0, stream>>>(ff_w1,  ff1_b, 1048576);
  cvt_bf16_kernel<<<4096, 256, 0, stream>>>(ff_w2,  ff2_b, 1048576);
  cvt_bf16_kernel<<<4096, 256, 0, stream>>>(dec_inp, din_b, 1048576);
  cvt_bf16_kernel<<<1024, 256, 0, stream>>>(r_emb,  remb_b, 262144);
  // rel-shift correction: corr[n][c] = rwb . r_emb[c]
  corr_kernel<<<64, 256, 0, stream>>>(r_w_bias, r_emb, corr);

  // heads = dec_inp @ qkv_w^T   (M=4096, N=3072, K=1024), bf16 out
  gemm_bb<false, false, true><<<dim3(12, 32), 256, 0, stream>>>(
      din_b, qkv_b, nullptr, headsb, 4096, 3072, 1024, 1024, 1024, 3072);
  // attention (per batch*head) -> attn_vec bf16, rows (i*BSZ+b), cols (n*64+d)
  flash_rel_attn<<<dim3(16, 64), 128, 0, stream>>>(headsb, remb_b, r_w_bias, r_bias, corr, avecb);
  // attn_out = attn_vec @ o_w^T  (M=4096, N=1024, K=1024), f32 out
  gemm_bb<false, false, false><<<dim3(4, 32), 256, 0, stream>>>(
      avecb, ow_b, nullptr, aout, 4096, 1024, 1024, 1024, 1024, 1024);
  // out1 = LN(dec_inp + attn_out), f32 + bf16 mirror
  ln_residual<true><<<4096, 256, 0, stream>>>(dec_inp, aout, ln1_g, ln1_b, out1, out1b);
  // h = relu(out1 @ ff_w1^T + b1) -> bf16  (M=4096, N=4096, K=1024)
  gemm_bb<true, true, true><<<dim3(16, 32), 256, 0, stream>>>(
      out1b, ff1_b, ff_b1, hbuf, 4096, 4096, 1024, 1024, 1024, 4096);
  // core = h @ ff_w2^T + b2  (M=4096, N=1024, K=4096), f32 out
  gemm_bb<true, false, false><<<dim3(4, 32), 256, 0, stream>>>(
      hbuf, ff2_b, ff_b2, core, 4096, 1024, 4096, 4096, 4096, 1024);
  // out = LN(out1 + core)
  ln_residual<false><<<4096, 256, 0, stream>>>(out1, core, ln2_g, ln2_b, (float*)d_out, nullptr);
}
